// MemoryBank_61323543052638
// MI455X (gfx1250) — matrix operand order repaired and checked by reading, not compiled, not run on hardware
//
#include <hip/hip_runtime.h>
#include <hip/hip_bf16.h>
#include <stddef.h>

// ---------------------------------------------------------------------------
// MemoryBank top-k retrieval for MI455X (gfx1250, wave32, WMMA).
//   q[1024,128] f32, keys[131072,128] f32, values[131072,256] f32, topk=5
// Pipeline:
//   K1: per-row L2 normalize + f32->f16 convert (q16, k16 into d_ws)
//   K2: WMMA f32_16x16x32_f16 cosine GEMM, double-buffered async global->LDS
//       key staging, fused per-(lane,v) top-5
//   K3: per-row candidate reduce (64 slices x 5 -> top-5), gather values, mean
// ---------------------------------------------------------------------------

typedef __attribute__((ext_vector_type(16))) _Float16 v16h;
typedef __attribute__((ext_vector_type(8)))  _Float16 v8h;
typedef __attribute__((ext_vector_type(4)))  _Float16 v4h;
typedef __attribute__((ext_vector_type(8)))  float    v8f;
typedef __attribute__((ext_vector_type(4)))  int      v4i;

#define NQ        1024
#define KD        128
#define VD        256
#define CAP       131072
#define TOPK      5
#define NSLICE    64                 // key slices (grid.x of K2)
#define SLICE_KEYS (CAP / NSLICE)    // 2048
#define CHUNK     128                // keys staged in LDS per iteration
#define NCHUNK    (SLICE_KEYS / CHUNK) // 16
#define ROWS_BLK  64                 // query rows per block (4 waves x 16)
#define LSTR      136                // LDS halves per row (128 + 8 pad)
#define KCHUNK_BYTES (CHUNK * LSTR * 2)   // 34816

// workspace layout (bytes)
#define Q16_BYTES ((size_t)NQ * KD * 2)             // 262144
#define K16_BYTES ((size_t)CAP * KD * 2)            // 33554432
#define CAND_N    (NSLICE * TOPK)                   // 320 candidates / row
#define CANDS_BYTES ((size_t)NQ * CAND_N * 4)

#if defined(__has_builtin)
#if __has_builtin(__builtin_amdgcn_global_load_async_to_lds_b128)
#define USE_ASYNC_LDS 1
#endif
#endif

__device__ __forceinline__ void wait_async0() {
#if defined(USE_ASYNC_LDS)
#if __has_builtin(__builtin_amdgcn_s_wait_asynccnt)
    __builtin_amdgcn_s_wait_asynccnt(0);
#else
    asm volatile("s_wait_asynccnt 0" ::: "memory");
#endif
#endif
}

// -------------------- small register-resident top-5 ------------------------
struct Top5 {
    float    s0, s1, s2, s3, s4;   // ascending: s0 = current 5th best
    unsigned i0, i1, i2, i3, i4;
};

__device__ inline void t5_init(Top5& t) {
    t.s0 = t.s1 = t.s2 = t.s3 = t.s4 = -2.0f;   // cosine >= -1
    t.i0 = t.i1 = t.i2 = t.i3 = t.i4 = 0u;
}

__device__ inline void t5_insert(Top5& t, float sc, unsigned id) {
    if (sc <= t.s0) return;
    if (sc > t.s1) {
        t.s0 = t.s1; t.i0 = t.i1;
        if (sc > t.s2) {
            t.s1 = t.s2; t.i1 = t.i2;
            if (sc > t.s3) {
                t.s2 = t.s3; t.i2 = t.i3;
                if (sc > t.s4) {
                    t.s3 = t.s4; t.i3 = t.i4; t.s4 = sc; t.i4 = id;
                } else { t.s3 = sc; t.i3 = id; }
            } else { t.s2 = sc; t.i2 = id; }
        } else { t.s1 = sc; t.i1 = id; }
    } else { t.s0 = sc; t.i0 = id; }
}

// -------------------- K1: normalize + f16 convert --------------------------
__global__ __launch_bounds__(256) void norm_f16_kernel(
    const float* __restrict__ q, const float* __restrict__ keys,
    _Float16* __restrict__ q16, _Float16* __restrict__ k16) {
    const int wave = threadIdx.x >> 5;
    const int lane = threadIdx.x & 31;
    const int row  = blockIdx.x * 8 + wave;         // 132096 rows total

    const float* src;
    _Float16* dst;
    if (row < NQ) { src = q + (size_t)row * KD;           dst = q16 + (size_t)row * KD; }
    else          { src = keys + (size_t)(row - NQ) * KD; dst = k16 + (size_t)(row - NQ) * KD; }

    const float4 v = ((const float4*)src)[lane];
    float ss = v.x * v.x + v.y * v.y + v.z * v.z + v.w * v.w;
    #pragma unroll
    for (int m = 16; m >= 1; m >>= 1) ss += __shfl_xor(ss, m, 32);
    const float inv = 1.0f / fmaxf(sqrtf(ss), 1e-8f);

    v4h o;
    o.x = (_Float16)(v.x * inv);
    o.y = (_Float16)(v.y * inv);
    o.z = (_Float16)(v.z * inv);
    o.w = (_Float16)(v.w * inv);
    *(v4h*)(dst + lane * 4) = o;
}

// -------------------- K2: WMMA scores + fused top-5 ------------------------
union AFrag { v16h v; v8h h[2]; };
struct Pair { float s; unsigned i; };

__global__ __launch_bounds__(128) void score_topk_kernel(
    const _Float16* __restrict__ q16, const _Float16* __restrict__ k16,
    float* __restrict__ candS, unsigned* __restrict__ candI) {

    __shared__ __align__(16) _Float16 qs[ROWS_BLK * LSTR];   // 17408 B
    __shared__ __align__(16) char     kbuf[2 * KCHUNK_BYTES]; // 69632 B (merge area reuses it)
    Pair* mp = (Pair*)kbuf;

    const int tid   = threadIdx.x;
    const int wave  = tid >> 5;
    const int lane  = tid & 31;
    const int l16   = lane & 15;         // N column / M row-within-half
    const int h     = lane >> 4;         // lane half
    const int slice = blockIdx.x;
    const int rowg  = blockIdx.y * ROWS_BLK;
    const unsigned sliceBase = (unsigned)slice * SLICE_KEYS;

    // --- stage 64x128 q16 tile: each thread copies half a row (128 B) ---
    {
        const int r = tid >> 1, part = tid & 1;
        const _Float16* g = q16 + (size_t)(rowg + r) * KD + part * 64;
        _Float16* l = qs + r * LSTR + part * 64;
        #pragma unroll
        for (int i = 0; i < 8; ++i) *(v8h*)(l + i * 8) = *(const v8h*)(g + i * 8);
    }
    __syncthreads();

    // --- per-wave A fragments (rows wave*16 .. +15), all 4 K-chunks ---
    // ISA 16-bit A 16x32 layout: lane(l16,h); halves [h*8, h*8+7] then [16+h*8 ..].
    AFrag A[4];
    {
        const _Float16* qrow = qs + (wave * 16 + l16) * LSTR;
        #pragma unroll
        for (int kb = 0; kb < 4; ++kb) {
            A[kb].h[0] = *(const v8h*)(qrow + kb * 32 + h * 8);
            A[kb].h[1] = *(const v8h*)(qrow + kb * 32 + 16 + h * 8);
        }
    }

    Top5 lists[8];
    #pragma unroll
    for (int v = 0; v < 8; ++v) t5_init(lists[v]);

    // async stage of one CHUNK of key rows into LDS buffer `dst` (no wait)
    auto stage = [&](int ch, char* dstBuf) {
        const unsigned krow = sliceBase + ch * CHUNK + tid;
        const _Float16* gsrc = k16 + (size_t)krow * KD;
        _Float16* ldst = (_Float16*)dstBuf + tid * LSTR;
#ifdef USE_ASYNC_LDS
        #pragma unroll
        for (int i = 0; i < 16; ++i)
            __builtin_amdgcn_global_load_async_to_lds_b128(
                (v4i*)(gsrc + i * 8), (v4i*)(ldst + i * 8), 0, 0);
#else
        #pragma unroll
        for (int i = 0; i < 16; ++i)
            *(v8h*)(ldst + i * 8) = *(const v8h*)(gsrc + i * 8);
#endif
    };

    // --- double-buffered key streaming: DMA chunk ch+1 while WMMA-ing ch ---
    stage(0, kbuf);
    wait_async0();
    __syncthreads();

    for (int ch = 0; ch < NCHUNK; ++ch) {
        const _Float16* cur = (const _Float16*)(kbuf + (size_t)(ch & 1) * KCHUNK_BYTES);
        const bool more = (ch + 1 < NCHUNK);
        if (more) stage(ch + 1, kbuf + (size_t)((ch + 1) & 1) * KCHUNK_BYTES);
        if (ch + 2 < NCHUNK) {   // pull chunk ch+2 toward L2 while we compute
            const _Float16* np = k16 + (size_t)(sliceBase + (ch + 2) * CHUNK + tid) * KD;
            __builtin_prefetch((const void*)np, 0, 1);
            __builtin_prefetch((const void*)(np + 64), 0, 1);
        }

        // 8 tiles of 16 keys; every wave scores its 16 rows vs all staged keys
        for (int nb = 0; nb < CHUNK; nb += 16) {
            const _Float16* krow = cur + (nb + l16) * LSTR;  // B column n = l16
            // preload ALL 4 B fragments first so the WMMA chain isn't
            // serialized on per-fragment s_wait_dscnt.
            AFrag B[4];
            #pragma unroll
            for (int kb = 0; kb < 4; ++kb) {
                // ISA 16-bit B 32x16 layout: lane(l16,h); halves [h*8, h*8+7] then [16+h*8 ..].
                B[kb].h[0] = *(const v8h*)(krow + kb * 32 + h * 8);
                B[kb].h[1] = *(const v8h*)(krow + kb * 32 + 16 + h * 8);
            }
            v8f acc = {};
            #pragma unroll
            for (int kb = 0; kb < 4; ++kb)
                acc = __builtin_amdgcn_wmma_f32_16x16x32_f16(
                    false, A[kb].v, false, B[kb].v, (short)0, acc, false, false);

            // C layout: VGPR v -> row m = v + 8*h ; column n = l16
            const unsigned key = sliceBase + (unsigned)(ch * CHUNK + nb) + l16;
            #pragma unroll
            for (int v = 0; v < 8; ++v) t5_insert(lists[v], acc[v], key);
        }

        if (more) wait_async0();   // own wave's DMA for ch+1 done
        __syncthreads();           // everyone done reading cur / next chunk visible
    }

    // --- merge per-lane lists -> per-row top-5 (reuse kbuf as Pair area) ---
    #pragma unroll
    for (int v = 0; v < 8; ++v) {
        const int rw = wave * 16 + 8 * h + v;       // block-local row of list v
        Pair* p = mp + (rw * 16 + l16) * TOPK;
        p[0] = {lists[v].s0, lists[v].i0};
        p[1] = {lists[v].s1, lists[v].i1};
        p[2] = {lists[v].s2, lists[v].i2};
        p[3] = {lists[v].s3, lists[v].i3};
        p[4] = {lists[v].s4, lists[v].i4};
    }
    __syncthreads();

    if (tid < ROWS_BLK) {
        Top5 t; t5_init(t);
        const Pair* p = mp + tid * (16 * TOPK);
        for (int c = 0; c < 16 * TOPK; ++c) t5_insert(t, p[c].s, p[c].i);
        const size_t base = ((size_t)(rowg + tid) * NSLICE + slice) * TOPK;
        candS[base + 0] = t.s0; candI[base + 0] = t.i0;
        candS[base + 1] = t.s1; candI[base + 1] = t.i1;
        candS[base + 2] = t.s2; candI[base + 2] = t.i2;
        candS[base + 3] = t.s3; candI[base + 3] = t.i3;
        candS[base + 4] = t.s4; candI[base + 4] = t.i4;
    }
}

// -------------------- K3: final top-5 + gather/mean ------------------------
__global__ __launch_bounds__(256) void fuse_kernel(
    const float* __restrict__ values, const float* __restrict__ candS,
    const unsigned* __restrict__ candI, float* __restrict__ out) {

    __shared__ float    ss[CAND_N];
    __shared__ unsigned si[CAND_N];
    __shared__ unsigned top[TOPK];

    const int row = blockIdx.x;
    for (int c = threadIdx.x; c < CAND_N; c += 256) {
        ss[c] = candS[(size_t)row * CAND_N + c];
        si[c] = candI[(size_t)row * CAND_N + c];
    }
    __syncthreads();

    if (threadIdx.x == 0) {
        Top5 t; t5_init(t);
        for (int c = 0; c < CAND_N; ++c) t5_insert(t, ss[c], si[c]);
        top[0] = t.i0; top[1] = t.i1; top[2] = t.i2; top[3] = t.i3; top[4] = t.i4;
    }
    __syncthreads();

    float acc = 0.0f;
    #pragma unroll
    for (int j = 0; j < TOPK; ++j)
        acc += values[(size_t)top[j] * VD + threadIdx.x];
    out[(size_t)row * VD + threadIdx.x] = acc * (1.0f / TOPK);
}

// ---------------------------------------------------------------------------
extern "C" void kernel_launch(void* const* d_in, const int* in_sizes, int n_in,
                              void* d_out, int out_size, void* d_ws, size_t ws_size,
                              hipStream_t stream) {
    (void)in_sizes; (void)n_in; (void)out_size; (void)ws_size;
    const float* q      = (const float*)d_in[0];
    const float* keys   = (const float*)d_in[1];
    const float* values = (const float*)d_in[2];
    // d_in[3] = topk (==5), compile-time constant here

    char* w = (char*)d_ws;
    _Float16* q16   = (_Float16*)(w);
    _Float16* k16   = (_Float16*)(w + Q16_BYTES);
    float*    candS = (float*)(w + Q16_BYTES + K16_BYTES);
    unsigned* candI = (unsigned*)(w + Q16_BYTES + K16_BYTES + CANDS_BYTES);

    // K1: 132096 rows, 8 waves/block
    norm_f16_kernel<<<(NQ + CAP) / 8, 256, 0, stream>>>(q, keys, q16, k16);
    // K2: 64 slices x 16 row-groups, 4 waves/block
    score_topk_kernel<<<dim3(NSLICE, NQ / ROWS_BLK), 128, 0, stream>>>(q16, k16, candS, candI);
    // K3: one block per query row
    fuse_kernel<<<NQ, VD, 0, stream>>>(values, candS, candI, (float*)d_out);
}
